// GCN_25142738550938
// MI455X (gfx1250) — matrix operand order repaired and checked by reading, not compiled, not run on hardware
//
#include <hip/hip_runtime.h>
#include <hip/hip_bf16.h>
#include <math.h>

// ---------------- problem constants (match reference) ----------------
#define NN 131072          // nodes
#define NE 524288          // edges
#define NB 4096            // graphs / batch
#define NPG 32             // nodes per graph
#define F 64               // GCN hidden feature width
#define L 768              // sequence length

typedef __attribute__((ext_vector_type(16))) _Float16 v16h;
typedef __attribute__((ext_vector_type(8)))  _Float16 v8h;
typedef __attribute__((ext_vector_type(8)))  float    v8f;

__device__ __forceinline__ float softplus_f(float x) {
  return fmaxf(x, 0.0f) + log1pf(expf(-fabsf(x)));
}
__device__ __forceinline__ float sigmoid_f(float x) {
  return 1.0f / (1.0f + expf(-x));
}

// ---------------- GCN degree / normalization ----------------
__global__ void k_fill1(float* p, int n) {
  int t = blockIdx.x * blockDim.x + threadIdx.x;
  if (t < n) p[t] = 1.0f;
}
__global__ void k_degcount(float* deg, const int* dst, int e) {
  int t = blockIdx.x * blockDim.x + threadIdx.x;
  if (t < e) atomicAdd(&deg[dst[t]], 1.0f);
}
__global__ void k_rsqrt_inplace(float* p, int n) {
  int t = blockIdx.x * blockDim.x + threadIdx.x;
  if (t < n) p[t] = rsqrtf(p[t]);
}

// ---------------- layer-1 GEMM: [N,9] x [9,64] (K too small for WMMA) ----------------
__global__ void k_gemm9(const float* __restrict__ x, const float* __restrict__ W,
                        float* __restrict__ h) {
  int t = blockIdx.x * blockDim.x + threadIdx.x;   // over NN*F
  int node = t >> 6, f = t & 63;
  float s = 0.0f;
  const float* xr = x + node * 9;
  #pragma unroll
  for (int i = 0; i < 9; ++i) s += xr[i] * W[i * F + f];
  h[t] = s;
}

// ---------------- WMMA GEMM: C[M,Nn] = A[M,K] * B[K,Nn] (+bias) ----------------
// Block = 128 threads (4 waves) covering a 16-row x 64-col output strip.
// A tile (16xK) staged in LDS as f16 row-major; B strip (Kx64) staged TRANSPOSED
// as f16 [64 cols][K] so every lane's fragment is contiguous ds_load_b128s.
// Compile-time K fully unrolls the k-loop (K/32 v_wmma per kernel).
template <int K>
__global__ void k_wmma_gemm_t(const float* __restrict__ A, const float* __restrict__ B,
                              const float* __restrict__ bias, float* __restrict__ C,
                              int M, int Nn) {
  __shared__ _Float16 sAh[16 * K];   // [row][k]
  __shared__ _Float16 sBt[64 * K];   // [col][k]  (transposed strip)

  const int tid  = threadIdx.x;
  const int wave = tid >> 5;
  const int lane = tid & 31;
  const int colBlocks = Nn >> 6;                 // strips of 64 columns
  const int rows0 = (blockIdx.x / colBlocks) << 4;
  const int cols0 = (blockIdx.x % colBlocks) << 6;

  // stage A tile: coalesced global reads (consecutive tid -> consecutive k)
  #pragma unroll
  for (int idx = tid; idx < 16 * K; idx += 128) {
    int r = idx / K, k = idx - r * K;
    sAh[idx] = (_Float16)A[(size_t)(rows0 + r) * K + k];
  }
  // stage B strip transposed: coalesced reads along Nn, write [col][k]
  #pragma unroll
  for (int idx = tid; idx < 64 * K; idx += 128) {
    int k = idx >> 6, n = idx & 63;              // consecutive tid -> consecutive n
    sBt[n * K + k] = (_Float16)B[(size_t)k * Nn + cols0 + n];
  }
  __syncthreads();

  const int m  = lane & 15;
  const int kh = lane >> 4;
  const int nloc = (wave << 4) + (lane & 15);    // this wave's column within strip

  v8f acc = {};
  #pragma unroll
  for (int k0 = 0; k0 < K; k0 += 32) {
    // A fragment: two contiguous 8-half runs of row m (ISA 7.12.2 layout)
    v8h f0 = *(const v8h*)(sAh + m * K + k0 + 8 * kh);
    v8h f1 = *(const v8h*)(sAh + m * K + k0 + 16 + 8 * kh);
    // B fragment: two contiguous 8-half runs of column nloc (same k layout as A)
    v8h g0 = *(const v8h*)(sBt + nloc * K + k0 + 8 * kh);
    v8h g1 = *(const v8h*)(sBt + nloc * K + k0 + 16 + 8 * kh);
    v16h a, b;
    #pragma unroll
    for (int j = 0; j < 8; ++j) {
      a[j] = f0[j]; a[j + 8] = f1[j];
      b[j] = g0[j]; b[j + 8] = g1[j];
    }
    acc = __builtin_amdgcn_wmma_f32_16x16x32_f16(false, a, false, b, (short)0, acc,
                                                 false, false);
  }

  // C/D layout: VGPR r -> M = r + 8*kh, N = lane&15 (within this wave's 16-col tile)
  const int col = cols0 + nloc;
  float bv = bias ? bias[col] : 0.0f;
  #pragma unroll
  for (int r = 0; r < 8; ++r)
    C[(size_t)(rows0 + r + 8 * kh) * Nn + col] = acc[r] + bv;
}

// ---------------- GCN aggregation ----------------
__global__ void k_selfinit(const float* __restrict__ h, const float* __restrict__ dinv,
                           float* __restrict__ acc) {
  int t = blockIdx.x * blockDim.x + threadIdx.x;   // over NN*F
  int node = t >> 6;
  float d = dinv[node];
  acc[t] = h[t] * d * d;
}
__global__ void k_edgeagg(const float* __restrict__ h, const float* __restrict__ dinv,
                          const int* __restrict__ src, const int* __restrict__ dst,
                          float* __restrict__ acc) {
  int t = blockIdx.x * blockDim.x + threadIdx.x;   // over NE*16
  int e = t >> 4, q = t & 15;
  int s = src[e], d = dst[e];
  float nrm = dinv[s] * dinv[d];
  int f0 = q * 4;
  const float* hs = h + (size_t)s * F + f0;
  float* ad = acc + (size_t)d * F + f0;
  #pragma unroll
  for (int j = 0; j < 4; ++j) atomicAdd(&ad[j], hs[j] * nrm);
}
__global__ void k_bias_act(float* __restrict__ acc, const float* __restrict__ bias,
                           int doTanh) {
  int t = blockIdx.x * blockDim.x + threadIdx.x;   // over NN*F
  float v = acc[t] + bias[t & 63];
  acc[t] = doTanh ? tanhf(v) : v;
}

// ---------------- graph pooling: max+mean over 32 nodes ----------------
__global__ void k_pool(const float* __restrict__ x, float* __restrict__ pooled) {
  int g = blockIdx.x;        // graph
  int f = threadIdx.x;       // 64 threads, one feature each
  const float* base = x + (size_t)g * NPG * F + f;
  float mx = -3.0e38f, sm = 0.0f;
  #pragma unroll 4
  for (int i = 0; i < NPG; ++i) {
    float v = base[i * F];
    mx = fmaxf(mx, v);
    sm += v;
  }
  pooled[g * 128 + f]      = mx;
  pooled[g * 128 + 64 + f] = sm * (1.0f / NPG);
}

// ---------------- fused CNN / CBAM / MLP tail: one block per batch element ----------------
// dynamic LDS layout (floats):
//   sComb [2*L]     conv1 input, later reused as SAM mean/max rows
//   sA    [32*L]    ping
//   sB    [32*L]    pong (later: spatial attention row, then pooled 12288 vector)
//   sW    [3072]    staged conv weights (conv2 then conv3)
//   sMisc [128]     CBAM stats / MLP temporaries
#define SM_FLOATS (2*L + 32*L + 32*L + 3072 + 128)

__device__ __forceinline__ void conv32x32(const float* __restrict__ in,
                                          float* __restrict__ out,
                                          const float* __restrict__ sW,
                                          const float* __restrict__ bias, int tid) {
  for (int idx = tid; idx < 32 * L; idx += 256) {
    int co = idx / L, l = idx - co * L;
    float sum = bias[co];
    const float* wco = sW + co * 96;
    #pragma unroll 4
    for (int ci = 0; ci < 32; ++ci) {
      const float* row = in + ci * L;
      float vm = (l > 0)     ? row[l - 1] : 0.0f;
      float v0 = row[l];
      float vp = (l < L - 1) ? row[l + 1] : 0.0f;
      const float* w = wco + ci * 3;
      sum += vm * w[0] + v0 * w[1] + vp * w[2];
    }
    out[idx] = sum;
  }
}

__global__ void k_cnn_tail(const float* __restrict__ gout,    // [B,768]
                           const float* __restrict__ protein, // [B,768]
                           const float* __restrict__ c1W, const float* __restrict__ c1b,
                           const float* __restrict__ c2W, const float* __restrict__ c2b,
                           const float* __restrict__ c3W, const float* __restrict__ c3b,
                           const float* __restrict__ caW1, const float* __restrict__ caW2,
                           const float* __restrict__ samW, const float* __restrict__ samb,
                           const float* __restrict__ l1W, const float* __restrict__ l1b,
                           const float* __restrict__ l2W, const float* __restrict__ l2b,
                           const float* __restrict__ l3W, const float* __restrict__ l3b,
                           float* __restrict__ out) {
  extern __shared__ float smem[];
  float* sComb = smem;                 // 2*L
  float* sA    = smem + 2 * L;         // 32*L
  float* sB    = sA + 32 * L;          // 32*L
  float* sW    = sB + 32 * L;          // 3072
  float* sMisc = sW + 3072;            // 128

  const int b = blockIdx.x;
  const int tid = threadIdx.x;

  // stage combined input [2,768]
  for (int l = tid; l < L; l += 256) {
    sComb[l]     = gout[(size_t)b * L + l];
    sComb[L + l] = protein[(size_t)b * L + l];
  }
  __syncthreads();

  // conv1: 2 -> 32, k=3, pad=1
  for (int idx = tid; idx < 32 * L; idx += 256) {
    int co = idx / L, l = idx - co * L;
    float sum = c1b[co];
    #pragma unroll
    for (int ci = 0; ci < 2; ++ci) {
      const float* row = sComb + ci * L;
      float vm = (l > 0)     ? row[l - 1] : 0.0f;
      float v0 = row[l];
      float vp = (l < L - 1) ? row[l + 1] : 0.0f;
      const float* w = c1W + (co * 2 + ci) * 3;
      sum += vm * w[0] + v0 * w[1] + vp * w[2];
    }
    sA[idx] = sum;
  }
  // stage conv2 weights
  for (int i = tid; i < 3072; i += 256) sW[i] = c2W[i];
  __syncthreads();

  conv32x32(sA, sB, sW, c2b, tid);     // conv2: sA -> sB
  __syncthreads();
  for (int i = tid; i < 3072; i += 256) sW[i] = c3W[i];
  __syncthreads();
  conv32x32(sB, sA, sW, c3b, tid);     // conv3: sB -> sA
  __syncthreads();

  // ---- CBAM channel attention ----
  if (tid < 32) {
    float mx = -3.0e38f, sm = 0.0f;
    const float* row = sA + tid * L;
    for (int l = 0; l < L; ++l) { float v = row[l]; mx = fmaxf(mx, v); sm += v; }
    sMisc[tid]      = mx;              // x1 (max)
    sMisc[32 + tid] = sm * (1.0f / L); // x2 (mean)
  }
  __syncthreads();
  if (tid < 16) {
    float h1 = 0.0f, h2 = 0.0f;
    #pragma unroll 4
    for (int c = 0; c < 32; ++c) {
      float w = caW1[tid * 32 + c];
      h1 += sMisc[c] * w;
      h2 += sMisc[32 + c] * w;
    }
    sMisc[64 + tid] = h1;
    sMisc[80 + tid] = h2;
  }
  __syncthreads();
  if (tid < 32) {
    float o1 = 0.0f, o2 = 0.0f;
    #pragma unroll
    for (int j = 0; j < 16; ++j) {
      float w = caW2[tid * 16 + j];
      o1 += sMisc[64 + j] * w;
      o2 += sMisc[80 + j] * w;
    }
    sMisc[96 + tid] = sigmoid_f(fmaxf(o1, 0.0f) + fmaxf(o2, 0.0f));
  }
  __syncthreads();
  for (int idx = tid; idx < 32 * L; idx += 256) sA[idx] *= sMisc[96 + idx / L];
  __syncthreads();

  // ---- CBAM spatial attention ----
  for (int l = tid; l < L; l += 256) {
    float sm = 0.0f, mx = -3.0e38f;
    #pragma unroll 4
    for (int c = 0; c < 32; ++c) { float v = sA[c * L + l]; sm += v; mx = fmaxf(mx, v); }
    sComb[l]     = sm * (1.0f / 32.0f);  // mean row
    sComb[L + l] = mx;                   // max row
  }
  __syncthreads();
  for (int l = tid; l < L; l += 256) {
    float sum = samb[0];
    #pragma unroll
    for (int ci = 0; ci < 2; ++ci) {
      const float* row = sComb + ci * L;
      float vm = (l > 0)     ? row[l - 1] : 0.0f;
      float v0 = row[l];
      float vp = (l < L - 1) ? row[l + 1] : 0.0f;
      const float* w = samW + ci * 3;
      sum += vm * w[0] + v0 * w[1] + vp * w[2];
    }
    sB[l] = sigmoid_f(sum);
  }
  __syncthreads();
  for (int idx = tid; idx < 32 * L; idx += 256) sA[idx] *= sB[idx % L];
  __syncthreads();

  // ---- MaxPool1d(2) + flatten -> sB[0..12287] ----
  for (int idx = tid; idx < 32 * (L / 2); idx += 256) {
    int c = idx / (L / 2), j = idx - c * (L / 2);
    sB[idx] = fmaxf(sA[c * L + 2 * j], sA[c * L + 2 * j + 1]);
  }
  __syncthreads();

  // ---- lin1: 12288 -> 32, softplus (8 waves, 4 output rows each) ----
  {
    int wv = tid >> 5, ln = tid & 31;
    for (int k = wv; k < 32; k += 8) {
      const float* Wk = l1W + (size_t)k * 12288;
      float s = 0.0f;
      for (int i = ln; i < 12288; i += 32) {
        __builtin_prefetch(Wk + i + 512, 0, 0);
        s += sB[i] * Wk[i];
      }
      #pragma unroll
      for (int off = 16; off > 0; off >>= 1) s += __shfl_xor(s, off, 32);
      if (ln == 0) sMisc[k] = softplus_f(s + l1b[k]);
    }
  }
  __syncthreads();

  // ---- lin2: 32 -> 32, softplus ----
  if (tid < 32) {
    float s = l2b[tid];
    #pragma unroll
    for (int k = 0; k < 32; ++k) s += sMisc[k] * l2W[tid * 32 + k];
    sMisc[64 + tid] = softplus_f(s);
  }
  __syncthreads();

  // ---- lin3: 32 -> 1, softplus ----
  if (tid == 0) {
    float s = l3b[0];
    #pragma unroll
    for (int k = 0; k < 32; ++k) s += sMisc[64 + k] * l3W[k];
    out[b] = softplus_f(s);
  }
}

// ---------------- launch ----------------
extern "C" void kernel_launch(void* const* d_in, const int* in_sizes, int n_in,
                              void* d_out, int out_size, void* d_ws, size_t ws_size,
                              hipStream_t stream) {
  const float* drug    = (const float*)d_in[0];
  const int*   eidx    = (const int*)d_in[1];
  const int*   src     = eidx;
  const int*   dst     = eidx + NE;
  const float* protein = (const float*)d_in[3];
  const float* g1W = (const float*)d_in[4];  const float* g1b = (const float*)d_in[5];
  const float* g2W = (const float*)d_in[6];  const float* g2b = (const float*)d_in[7];
  const float* oW  = (const float*)d_in[8];  const float* ob  = (const float*)d_in[9];
  const float* c1W = (const float*)d_in[10]; const float* c1b = (const float*)d_in[11];
  const float* c2W = (const float*)d_in[12]; const float* c2b = (const float*)d_in[13];
  const float* c3W = (const float*)d_in[14]; const float* c3b = (const float*)d_in[15];
  const float* caW1= (const float*)d_in[16]; const float* caW2= (const float*)d_in[17];
  const float* samW= (const float*)d_in[18]; const float* samb= (const float*)d_in[19];
  const float* l1W = (const float*)d_in[20]; const float* l1b = (const float*)d_in[21];
  const float* l2W = (const float*)d_in[22]; const float* l2b = (const float*)d_in[23];
  const float* l3W = (const float*)d_in[24]; const float* l3b = (const float*)d_in[25];
  float* out = (float*)d_out;

  float* ws     = (float*)d_ws;
  float* dinv   = ws;                                 // NN
  float* hbuf   = dinv + NN;                          // NN*F
  float* accA   = hbuf + (size_t)NN * F;              // NN*F
  float* accB   = accA + (size_t)NN * F;              // NN*F
  float* pooled = accB + (size_t)NN * F;              // NB*128
  float* gout   = pooled + (size_t)NB * 128;          // NB*768

  const int T = 256;
  const int gNodes  = NN / T;               // 512
  const int gNF     = NN * F / T;           // 32768
  const int gEdges  = NE / T;               // 2048
  const int gEF     = NE * 16 / T;          // 32768

  // degree -> dinv
  k_fill1<<<gNodes, T, 0, stream>>>(dinv, NN);
  k_degcount<<<gEdges, T, 0, stream>>>(dinv, dst, NE);
  k_rsqrt_inplace<<<gNodes, T, 0, stream>>>(dinv, NN);

  // GCN layer 1: h = drug @ W1 ; aggregate ; tanh(+b1) -> accA
  k_gemm9<<<gNF, T, 0, stream>>>(drug, g1W, hbuf);
  k_selfinit<<<gNF, T, 0, stream>>>(hbuf, dinv, accA);
  k_edgeagg<<<gEF, T, 0, stream>>>(hbuf, dinv, src, dst, accA);
  k_bias_act<<<gNF, T, 0, stream>>>(accA, g1b, 1);

  // GCN layer 2 (WMMA GEMM, K=64): h = x1 @ W2
  k_wmma_gemm_t<64><<<(NN / 16) * (F / 64), 128, 0, stream>>>(accA, g2W, nullptr,
                                                              hbuf, NN, F);
  k_selfinit<<<gNF, T, 0, stream>>>(hbuf, dinv, accB);
  k_edgeagg<<<gEF, T, 0, stream>>>(hbuf, dinv, src, dst, accB);
  k_bias_act<<<gNF, T, 0, stream>>>(accB, g2b, 1);

  // GCN layer 3 (same weights, no activation)
  k_wmma_gemm_t<64><<<(NN / 16) * (F / 64), 128, 0, stream>>>(accB, g2W, nullptr,
                                                              hbuf, NN, F);
  k_selfinit<<<gNF, T, 0, stream>>>(hbuf, dinv, accA);
  k_edgeagg<<<gEF, T, 0, stream>>>(hbuf, dinv, src, dst, accA);
  k_bias_act<<<gNF, T, 0, stream>>>(accA, g2b, 0);

  // graph pooling: max+mean -> pooled [B,128]
  k_pool<<<NB, 64, 0, stream>>>(accA, pooled);

  // output layer (WMMA GEMM + bias, K=128): gout = pooled @ out_W + out_b  [B,768]
  k_wmma_gemm_t<128><<<(NB / 16) * (L / 64), 128, 0, stream>>>(pooled, oW, ob,
                                                               gout, NB, L);

  // fused CNN / CBAM / MLP tail, one block per batch element
  k_cnn_tail<<<NB, 256, SM_FLOATS * sizeof(float), stream>>>(
      gout, protein, c1W, c1b, c2W, c2b, c3W, c3b, caW1, caW2, samW, samb,
      l1W, l1b, l2W, l2b, l3W, l3b, out);
}
